// KernalAttention_36060545417824
// MI455X (gfx1250) — matrix operand order repaired and checked by reading, not compiled, not run on hardware
//
#include <hip/hip_runtime.h>
#include <hip/hip_bf16.h>

// Linear ("kernel") attention, f16 WMMA path for gfx1250 (MI455X).
// out[b,h,n,v] = (sum_m S[n,m] * V[m,v]) / S[n,v],  S = phi(Q) phi(K)^T, phi = elu+1.

#define BH        64          // B*H
#define NN        1024        // sequence length (and dv)
#define DD        64          // head dim
#define BLOCK_N   64          // output rows per workgroup
#define BLOCK_V   128         // output cols per workgroup
#define M_STEP    64          // m-chunk per iteration
#define NTHREADS  512         // 16 waves (wave32)
#define LDS_STRIDE 72         // halves per LDS row: 64 data + 8 pad (row = 144B, 16B-aligned)

typedef __attribute__((ext_vector_type(16))) _Float16 v16h;
typedef __attribute__((ext_vector_type(8)))  float    v8f;
typedef __attribute__((ext_vector_type(4)))  float    f32x4;

union Frag16 { v16h v; f32x4 f4[2]; };
union H8     { f32x4 f4; _Float16 h[8]; };

__device__ __forceinline__ float phi(float x) {
    // elu(x, alpha=1) + 1  ==  x+1 (x>0)  |  exp(x) (x<=0)
    return x > 0.0f ? x + 1.0f : __expf(x);
}

// A-fragment (16x32, f16) from row-major LDS tile [row][k], per ISA A layout:
// lanes 0-15: K = kbase+0..7 (elems 0-7), kbase+16..23 (elems 8-15); lanes 16-31: +8.
__device__ __forceinline__ v16h load_a(const _Float16* base, int row0, int kbase, int lane) {
    const int hi = lane >> 4;
    const _Float16* p = base + (row0 + (lane & 15)) * LDS_STRIDE + kbase + hi * 8;
    Frag16 fr;
    fr.f4[0] = *(const f32x4*)(p);        // 8 halves: K +0..7
    fr.f4[1] = *(const f32x4*)(p + 16);   // 8 halves: K +16..23
    return fr.v;
}

// B-fragment (32x16, f16) from row-major LDS tile [ncol][k], per ISA B layout:
// lanes 0-15: K = kbase+0..7 (elems 0-7), kbase+16..23 (elems 8-15); lanes 16-31: +8.
__device__ __forceinline__ v16h load_b(const _Float16* base, int row0, int kbase, int lane) {
    const int hi = lane >> 4;
    const _Float16* p = base + (row0 + (lane & 15)) * LDS_STRIDE + kbase + hi * 8;
    Frag16 fr;
    fr.f4[0] = *(const f32x4*)(p);        // K +0..7
    fr.f4[1] = *(const f32x4*)(p + 16);   // K +16..23
    return fr.v;
}

__device__ __forceinline__ v8f wmma16(v16h a, v16h b, v8f c) {
    return __builtin_amdgcn_wmma_f32_16x16x32_f16(false, a, false, b, (short)0, c,
                                                  false, false);
}

__device__ __forceinline__ void phi8_store(_Float16* dst, f32x4 x0, f32x4 x1) {
    H8 h;
    h.h[0] = (_Float16)phi(x0.x); h.h[1] = (_Float16)phi(x0.y);
    h.h[2] = (_Float16)phi(x0.z); h.h[3] = (_Float16)phi(x0.w);
    h.h[4] = (_Float16)phi(x1.x); h.h[5] = (_Float16)phi(x1.y);
    h.h[6] = (_Float16)phi(x1.z); h.h[7] = (_Float16)phi(x1.w);
    *(f32x4*)dst = h.f4;
}

__global__ __launch_bounds__(NTHREADS)
void KernalAttention_kernel(const float* __restrict__ Q,
                            const float* __restrict__ K,
                            const float* __restrict__ V,
                            float* __restrict__ Out) {
    __shared__ __align__(16) _Float16 sQ [BLOCK_N * LDS_STRIDE];  // phi(Q) tile   [n][d]
    __shared__ __align__(16) _Float16 sK [M_STEP  * LDS_STRIDE];  // phi(K) chunk  [m][d]
    __shared__ __align__(16) _Float16 sS [BLOCK_N * LDS_STRIDE];  // S chunk (f16) [n][m]
    __shared__ __align__(16) _Float16 sVt[BLOCK_V * LDS_STRIDE];  // V^T chunk     [v][m]

    const int tid  = threadIdx.x;
    const int lane = tid & 31;
    const int wave = tid >> 5;       // 0..15
    const int wi   = wave >> 2;      // n-tile index 0..3   (rows wi*16..)
    const int wj   = wave & 3;       // v-tile index 0..3   (handles v tiles wj and wj+4)

    const int bh = blockIdx.z;
    const int n0 = blockIdx.y * BLOCK_N;
    const int v0 = blockIdx.x * BLOCK_V;

    const float* Qb = Q   + (size_t)bh * NN * DD;
    const float* Kb = K   + (size_t)bh * NN * DD;
    const float* Vb = V   + (size_t)bh * NN * NN;
    float*       Ob = Out + (size_t)bh * NN * NN;

    // ---- phi(Q) tile -> LDS (once) : 64x64 f32 -> f16, 8 elems/thread ----
    {
        const int r = tid >> 3;            // 0..63
        const int c = (tid & 7) * 8;       // 0..56
        const float* src = Qb + (size_t)(n0 + r) * DD + c;
        f32x4 x0 = *(const f32x4*)(src);
        f32x4 x1 = *(const f32x4*)(src + 4);
        phi8_store(&sQ[r * LDS_STRIDE + c], x0, x1);
    }

    v8f num0 = {}, num1 = {};   // numerator accumulators for the wave's two v-tiles
    v8f den0 = {}, den1 = {};   // denominator tiles (f32, captured from S compute)

#pragma unroll 1
    for (int m0 = 0; m0 < NN; m0 += M_STEP) {
        __syncthreads();   // previous iteration finished reading sK/sS/sVt (also fences sQ on iter 0)

        // ---- phi(K) chunk -> LDS : 64x64 ----
        {
            const int r = tid >> 3;
            const int c = (tid & 7) * 8;
            const float* src = Kb + (size_t)(m0 + r) * DD + c;
            f32x4 x0 = *(const f32x4*)(src);
            f32x4 x1 = *(const f32x4*)(src + 4);
            phi8_store(&sK[r * LDS_STRIDE + c], x0, x1);
        }
        // ---- V chunk -> LDS transposed : V[m0+mr][v0+vc] -> sVt[vc][mr] (f16) ----
        {
            const int mr  = tid >> 3;            // 0..63
            const int vc0 = (tid & 7) * 16;      // 0..112
            const float* src = Vb + (size_t)(m0 + mr) * NN + v0 + vc0;
#pragma unroll
            for (int q = 0; q < 4; ++q) {
                f32x4 x = *(const f32x4*)(src + q * 4);
                sVt[(vc0 + q * 4 + 0) * LDS_STRIDE + mr] = (_Float16)x.x;
                sVt[(vc0 + q * 4 + 1) * LDS_STRIDE + mr] = (_Float16)x.y;
                sVt[(vc0 + q * 4 + 2) * LDS_STRIDE + mr] = (_Float16)x.z;
                sVt[(vc0 + q * 4 + 3) * LDS_STRIDE + mr] = (_Float16)x.w;
            }
        }
        __syncthreads();

        // ---- Stage A: S subtile (wi, wj) = phiQ[wi*16.., :] * phiK[wj*16.., :]^T ----
        {
            v16h aq0 = load_a(sQ, wi * 16, 0,  lane);
            v16h aq1 = load_a(sQ, wi * 16, 32, lane);
            v16h bk0 = load_b(sK, wj * 16, 0,  lane);   // B[d][m] = phiK[m][d]
            v16h bk1 = load_b(sK, wj * 16, 32, lane);
            v8f s = {};
            s = wmma16(aq0, bk0, s);
            s = wmma16(aq1, bk1, s);

            // denominator capture: this wave's v-tiles are exactly the subtile it computed
            if (m0 == v0)      den0 = s;
            if (m0 == v0 + 64) den1 = s;

            // C-layout -> row-major f16 in sS (performs the lane transpose via LDS)
            const int hi  = lane >> 4;
            const int col = wj * 16 + (lane & 15);
            const int rb  = wi * 16 + 8 * hi;
#pragma unroll
            for (int r = 0; r < 8; ++r)
                sS[(rb + r) * LDS_STRIDE + col] = (_Float16)s[r];
        }
        __syncthreads();

        // ---- Stage C: numerator += S[n-tile, m-chunk] * V[m-chunk, v-tiles] ----
        {
            v16h sa0 = load_a(sS, wi * 16, 0,  lane);
            v16h sa1 = load_a(sS, wi * 16, 32, lane);

            v16h vb0 = load_b(sVt, wj * 16, 0,  lane);
            v16h vb1 = load_b(sVt, wj * 16, 32, lane);
            num0 = wmma16(sa0, vb0, num0);
            num0 = wmma16(sa1, vb1, num0);

            v16h vc0 = load_b(sVt, 64 + wj * 16, 0,  lane);
            v16h vc1 = load_b(sVt, 64 + wj * 16, 32, lane);
            num1 = wmma16(sa0, vc0, num1);
            num1 = wmma16(sa1, vc1, num1);
        }
    }

    // ---- out = num / den, C layout: VGPR r -> row (r + 8*hi), col = lane&15 ----
    {
        const int hi  = lane >> 4;
        const int col = lane & 15;
        const int nr  = n0 + wi * 16 + 8 * hi;
        const int vA  = v0 + wj * 16 + col;
        const int vB  = vA + 64;
#pragma unroll
        for (int r = 0; r < 8; ++r) {
            Ob[(size_t)(nr + r) * NN + vA] = num0[r] / den0[r];
            Ob[(size_t)(nr + r) * NN + vB] = num1[r] / den1[r];
        }
    }
}

extern "C" void kernel_launch(void* const* d_in, const int* in_sizes, int n_in,
                              void* d_out, int out_size, void* d_ws, size_t ws_size,
                              hipStream_t stream) {
    (void)in_sizes; (void)n_in; (void)out_size; (void)d_ws; (void)ws_size;
    const float* Q = (const float*)d_in[0];
    const float* K = (const float*)d_in[1];
    const float* V = (const float*)d_in[2];
    float* Out = (float*)d_out;

    dim3 grid(NN / BLOCK_V, NN / BLOCK_N, BH);   // (8, 16, 64)
    dim3 block(NTHREADS);                        // 512 = 16 waves
    KernalAttention_kernel<<<grid, block, 0, stream>>>(Q, K, V, Out);
}
